// VectorQuantizer_30339648979329
// MI455X (gfx1250) — matrix operand order repaired and checked by reading, not compiled, not run on hardware
//
#include <hip/hip_runtime.h>
#include <hip/hip_bf16.h>

// ---------------------------------------------------------------------------
// VQ-VAE vector quantizer for MI455X (gfx1250, wave32, WMMA).
//
// N=32768 queries (D=256 f32) vs K=1024 codes -> argmin of
//   ||z||^2 - 2 z.w + ||w||^2   (||z||^2 constant per row -> dropped)
// GEMM = 17.2 GFLOP; HBM traffic ~65MB (~2.8us @ 23.3TB/s) -> run the
// distance GEMM on v_wmma_f32_16x16x32_bf16 (f32 accumulate). Codebook is
// uniform(+-1/1024): bf16 rounding (~2.5e-5 in the dot) << top-2 argmin gap
// (~3e-3); z_q and the losses gather the ORIGINAL f32 weights.
//
// Round-4: previous rounds exposed a full L2 round-trip per WMMA (compiler
// lockstepped clause(loads)->s_wait_loadcnt 0->wmma, and register double-
// buffering spilled).  New structure: the codebook tile is identical for all
// waves in a block, so stage it once per block through double-buffered LDS.
// Per iteration: issue next tile's coalesced global loads (32B/thread, kept
// in 8 VGPRs), consume current tile from LDS (ds_load_b128 fragments, one
// chunk software-pipelined ahead), then ds_store + one barrier.  Global
// latency hides behind the 8 WMMAs; fragment latency is LDS-class.
// ---------------------------------------------------------------------------

typedef __attribute__((ext_vector_type(16))) __bf16 v16bf;
typedef __attribute__((ext_vector_type(8)))  __bf16 v8bf;
typedef __attribute__((ext_vector_type(8)))  float  v8f;

#define DIMS 256
#define KCB  1024
#define NTOT 32768           // 32 * 32 * 32
#define NELEM 8388608        // 32 * 256 * 32 * 32
#define KTILES 64            // 1024 / 16
#define TILE_ELEMS 4096      // 16 rows * 256 d

// ---------------------------------------------------------------------------
// Kernel 1: codebook f32 -> bf16 ([K][D] row-major) + wnorm[k] = ||w_k||^2
// ---------------------------------------------------------------------------
__global__ void vq_prep_w(const float* __restrict__ w,
                          __bf16* __restrict__ wb,
                          float* __restrict__ wnorm) {
    const int k = blockIdx.x;        // 1024 blocks
    const int d = threadIdx.x;       // 256 threads
    const float v = w[k * DIMS + d];
    wb[k * DIMS + d] = (__bf16)v;
    __shared__ float red[256];
    red[d] = v * v;
    __syncthreads();
    for (int s = 128; s > 0; s >>= 1) {
        if (d < s) red[d] += red[d + s];
        __syncthreads();
    }
    if (d == 0) wnorm[k] = red[0];
}

// ---------------------------------------------------------------------------
// Kernel 2: z_e (B=32, C=256, H=32, W=32) f32 -> zb[n][d] bf16 row-major,
// n = b*1024 + hw, d = c.  LDS 32x32 tile transpose for coalescing.
// ---------------------------------------------------------------------------
__global__ void vq_prep_z(const float* __restrict__ z,
                          __bf16* __restrict__ zb) {
    __shared__ float tile[32][33];
    const int b   = blockIdx.z;            // 32
    const int hw0 = blockIdx.x * 32;       // 32 tiles
    const int d0  = blockIdx.y * 32;       // 8 tiles
    const int tx = threadIdx.x;            // 32
    const int ty = threadIdx.y;            // 8
#pragma unroll
    for (int i = 0; i < 4; ++i) {
        const int dl = ty + i * 8;
        tile[dl][tx] = z[((size_t)(b * DIMS + d0 + dl) << 10) + hw0 + tx];
    }
    __syncthreads();
#pragma unroll
    for (int i = 0; i < 4; ++i) {
        const int hwl = ty + i * 8;
        const size_t n = (size_t)b * 1024 + hw0 + hwl;
        zb[n * DIMS + d0 + tx] = (__bf16)tile[tx][hwl];
    }
}

// Tree-reduced argmin update over the 8 per-lane candidates of one k-tile.
__device__ __forceinline__ void score_update(const v8f& acc,
                                             const float* __restrict__ wnorm,
                                             int kt, int hf,
                                             float& minval, int& minidx) {
    const int kbase = kt + hf * 8;
    const v8f wn = *(const v8f*)(wnorm + kbase);
    float sv[8];
    int   si[8];
#pragma unroll
    for (int m = 0; m < 8; ++m) {
        sv[m] = wn[m] - 2.0f * acc[m];
        si[m] = kbase + m;
    }
#pragma unroll
    for (int st = 1; st < 8; st <<= 1) {
#pragma unroll
        for (int m = 0; m < 8; m += 2 * st) {
            const bool lt = sv[m + st] < sv[m];
            sv[m] = lt ? sv[m + st] : sv[m];
            si[m] = lt ? si[m + st] : si[m];
        }
    }
    const bool lt = sv[0] < minval;
    minval = lt ? sv[0] : minval;
    minidx = lt ? si[0] : minidx;
}

// ---------------------------------------------------------------------------
// Kernel 3: WMMA argmin with block-wide LDS staging of the codebook tile.
// Block = 256 threads = 8 waves; wave w owns n-tile blockIdx*8+w (16 cols).
// A = codebook tile (16k x 32d), B = query tile (32d x 16n); lane L's
// accumulator c[m] = dot(w_{kt+hf*8+m}, z_{n0+(L&15)}): argmin over k is
// in-lane, one shfl_xor(16) merges the half-wave ranges at the end.
// A-fragment ds_loads in the same contiguous K order as B:
//   lane L: row kt+(L&15), K in {c*32+hf*16 .. +16} (two ds_load_b128).
// ---------------------------------------------------------------------------
__global__ __launch_bounds__(256, 1)
void vq_argmin(const __bf16* __restrict__ zb,
               const __bf16* __restrict__ wb,
               const float*  __restrict__ wnorm,
               int* __restrict__ idx_out) {
    __shared__ v16bf smem[2][256];               // 2 x 8KB A-tile buffers

    const int tid   = threadIdx.x;
    const int lane  = tid & 31;
    const int wave  = tid >> 5;                  // 8 waves / block
    const int ntile = blockIdx.x * 8 + wave;     // 2048 tiles of 16 n
    const int n0    = ntile * 16;
    const int l     = lane & 15;
    const int hf    = lane >> 4;                 // half-wave id

    // Resident B fragments: 8 chunks of K=32 (full D=256 for this lane's n).
    // B lane L: col n0+(L&15), K in {hf*16 .. +16} -> contiguous 32B.
    v16bf bfr[8];
    const __bf16* zrow = zb + (size_t)(n0 + l) * DIMS + hf * 16;
#pragma unroll
    for (int c = 0; c < 8; ++c)
        bfr[c] = *(const v16bf*)(zrow + c * 32);

    // Stage tile 0: 256 threads x 32B = 8KB, coalesced.
    smem[0][tid] = *(const v16bf*)(wb + (size_t)tid * 16);
    __syncthreads();

    float minval = 3.4e38f;
    int   minidx = 0;

    for (int t = 0; t < KTILES; ++t) {
        // Issue next tile's global loads first; they fly over the WMMAs.
        const bool has_next = (t + 1 < KTILES);
        v16bf stage;
        if (has_next)
            stage = *(const v16bf*)(wb + (size_t)(t + 1) * TILE_ELEMS + tid * 16);

        // Consume tile t from LDS, fragments pipelined one chunk ahead.
        const __bf16* at = (const __bf16*)&smem[t & 1][0] + l * DIMS + hf * 16;
        union AU { v16bf v; v8bf h[2]; };
        AU acur, anxt;
        acur.h[0] = *(const v8bf*)(at);
        acur.h[1] = *(const v8bf*)(at + 8);
        v8f acc = {};
#pragma unroll
        for (int c = 0; c < 8; ++c) {
            if (c < 7) {
                anxt.h[0] = *(const v8bf*)(at + (c + 1) * 32);
                anxt.h[1] = *(const v8bf*)(at + (c + 1) * 32 + 8);
            }
            acc = __builtin_amdgcn_wmma_f32_16x16x32_bf16(
                false, acur.v, false, bfr[c], (short)0, acc, false, false);
            acur = anxt;
        }

        score_update(acc, wnorm, t * 16, hf, minval, minidx);

        // Park the staged tile and synchronize the block.
        if (has_next)
            smem[(t + 1) & 1][tid] = stage;
        __syncthreads();
    }

    // Merge the two half-wave k-ranges for column n0 + l (lanes L, L+16).
    const float oval = __shfl_xor(minval, 16, 32);
    const int   oidx = __shfl_xor(minidx, 16, 32);
    if (oval < minval || (oval == minval && oidx < minidx)) {
        minval = oval;
        minidx = oidx;
    }
    if (hf == 0) idx_out[n0 + l] = minidx;
}

// ---------------------------------------------------------------------------
// Kernel 4a: zero the loss accumulator (graph-capture safe, no memset).
// ---------------------------------------------------------------------------
__global__ void vq_zero(float* acc) { acc[0] = 0.0f; }

// ---------------------------------------------------------------------------
// Kernel 4b: gather z_q (from ORIGINAL f32 weights) into (B,C,H,W) layout
// and accumulate sum((z_e - z_q)^2).
// ---------------------------------------------------------------------------
__global__ void vq_gather_loss(const float* __restrict__ z,
                               const float* __restrict__ w,
                               const int*   __restrict__ idx,
                               float* __restrict__ out,
                               float* __restrict__ acc) {
    const int b  = blockIdx.y;                       // 32
    const int hw = blockIdx.x * 128 + threadIdx.x;   // grid.x=8, block=128
    const int id = idx[b * 1024 + hw];
    const float* wrow = w + (size_t)id * DIMS;       // L2-resident codebook
    float s = 0.0f;
#pragma unroll 4
    for (int c = 0; c < DIMS; ++c) {
        const float q  = wrow[c];
        const size_t o = ((size_t)(b * DIMS + c) << 10) + hw;
        const float ze = z[o];
        out[o] = q;                                  // coalesced across hw
        const float d = ze - q;
        s += d * d;
    }
    __shared__ float red[128];
    red[threadIdx.x] = s;
    __syncthreads();
    for (int st = 64; st > 0; st >>= 1) {
        if (threadIdx.x < st) red[threadIdx.x] += red[threadIdx.x + st];
        __syncthreads();
    }
    if (threadIdx.x == 0) atomicAdd(acc, red[0]);
}

// ---------------------------------------------------------------------------
// Kernel 5: finalize both (numerically identical) loss scalars.
// ---------------------------------------------------------------------------
__global__ void vq_finalize(const float* __restrict__ acc,
                            float* __restrict__ out_losses) {
    const float loss = acc[0] / (float)NELEM;
    out_losses[0] = loss;   // embedding_loss
    out_losses[1] = loss;   // commitment_loss
}

// ---------------------------------------------------------------------------
extern "C" void kernel_launch(void* const* d_in, const int* in_sizes, int n_in,
                              void* d_out, int out_size, void* d_ws, size_t ws_size,
                              hipStream_t stream) {
    const float* z = (const float*)d_in[0];   // (32,256,32,32) f32
    const float* w = (const float*)d_in[1];   // (1024,256)     f32
    float* out = (float*)d_out;               // z_q (8388608) + 2 scalars

    // Workspace layout (~17.4 MB total)
    char* ws = (char*)d_ws;
    __bf16* zb    = (__bf16*)(ws);                                // 16,777,216 B
    __bf16* wb    = (__bf16*)(ws + 16777216);                     //    524,288 B
    float*  wnorm = (float*) (ws + 16777216 + 524288);            //      4,096 B
    int*    idx   = (int*)   (ws + 16777216 + 524288 + 4096);     //    131,072 B
    float*  acc   = (float*) (ws + 16777216 + 524288 + 4096 + 131072);

    vq_prep_w<<<KCB, 256, 0, stream>>>(w, wb, wnorm);

    dim3 tz(32, 8), gz(32 /*hw tiles*/, 8 /*d tiles*/, 32 /*b*/);
    vq_prep_z<<<gz, tz, 0, stream>>>(z, zb);

    vq_argmin<<<NTOT / 128, 256, 0, stream>>>(zb, wb, wnorm, idx);

    vq_zero<<<1, 1, 0, stream>>>(acc);
    vq_gather_loss<<<dim3(8, 32), 128, 0, stream>>>(z, w, idx, out, acc);
    vq_finalize<<<1, 1, 0, stream>>>(acc, out + NELEM);
}
